// DepthNet_60455959658989
// MI455X (gfx1250) — matrix operand order repaired and checked by reading, not compiled, not run on hardware
//
#include <hip/hip_runtime.h>
#include <hip/hip_bf16.h>

// Problem constants (from reference setup_inputs)
#define BB 2
#define VV 3
#define CC 32
#define HH 128
#define WW 160
#define DD 48
#define TD 4      // depth-tile of conv block
#define THT 4     // height-tile of conv block
#define HWN (HH*WW)
#define BHW (BB*HH*WW)
#define VOLN (BB*DD*HH*WW)       // 1,966,080 positions

typedef __attribute__((ext_vector_type(16))) _Float16 v16h;
typedef __attribute__((ext_vector_type(8)))  _Float16 v8h;
typedef __attribute__((ext_vector_type(8)))  float    v8f;
typedef __attribute__((ext_vector_type(4)))  unsigned int u32x4;
typedef __attribute__((ext_vector_type(8)))  int i32x8;
typedef __attribute__((ext_vector_type(4)))  int i32x4;

#if defined(__has_builtin)
#if __has_builtin(__builtin_amdgcn_tensor_load_to_lds) && __has_builtin(__builtin_amdgcn_s_wait_tensorcnt)
#define USE_TDM 1
#endif
#endif
#ifndef USE_TDM
#define USE_TDM 0
#endif

// ---------------------------------------------------------------------------
// Kernel 0: per (b, src view v in {1,2}) compute rot/trans of
//   proj = compose(src) @ inv(compose(ref)), store 12 floats each.
// ---------------------------------------------------------------------------
__device__ inline void mat4_compose(const float* E, const float* K, float* O) {
  // O = E; O[:3,:4] = K[:3,:3] @ E[:3,:4]
  for (int i = 0; i < 16; ++i) O[i] = E[i];
  for (int r = 0; r < 3; ++r)
    for (int c = 0; c < 4; ++c) {
      float s = 0.f;
      for (int j = 0; j < 3; ++j) s += K[r*4+j] * E[j*4+c];
      O[r*4+c] = s;
    }
}

__device__ inline void mat4_inv(const float* M, float* Inv) {
  // Gauss-Jordan with partial pivoting on [M | I]
  float a[4][8];
  for (int r = 0; r < 4; ++r)
    for (int c = 0; c < 4; ++c) { a[r][c] = M[r*4+c]; a[r][4+c] = (r == c) ? 1.f : 0.f; }
  for (int k = 0; k < 4; ++k) {
    int piv = k;
    float best = fabsf(a[k][k]);
    for (int r = k+1; r < 4; ++r) { float v = fabsf(a[r][k]); if (v > best) { best = v; piv = r; } }
    if (piv != k) for (int c = 0; c < 8; ++c) { float t = a[k][c]; a[k][c] = a[piv][c]; a[piv][c] = t; }
    float inv = 1.f / a[k][k];
    for (int c = 0; c < 8; ++c) a[k][c] *= inv;
    for (int r = 0; r < 4; ++r) {
      if (r == k) continue;
      float f = a[r][k];
      for (int c = 0; c < 8; ++c) a[r][c] -= f * a[k][c];
    }
  }
  for (int r = 0; r < 4; ++r)
    for (int c = 0; c < 4; ++c) Inv[r*4+c] = a[r][4+c];
}

__global__ void k_setup(const float* __restrict__ projm, float* __restrict__ pp) {
  int t = threadIdx.x;
  if (t >= BB * (VV - 1)) return;
  int b = t / (VV - 1);
  int v = (t % (VV - 1)) + 1;
  float E0[16], K0[16], Ev[16], Kv[16];
  for (int i = 0; i < 16; ++i) {
    E0[i] = projm[(((b*VV + 0)*2 + 0)*16) + i];
    K0[i] = projm[(((b*VV + 0)*2 + 1)*16) + i];
    Ev[i] = projm[(((b*VV + v)*2 + 0)*16) + i];
    Kv[i] = projm[(((b*VV + v)*2 + 1)*16) + i];
  }
  float Rn[16], Sn[16], Ri[16], P[16];
  mat4_compose(E0, K0, Rn);          // ref_proj_new
  mat4_compose(Ev, Kv, Sn);          // src_proj_new
  mat4_inv(Rn, Ri);
  for (int r = 0; r < 4; ++r)
    for (int c = 0; c < 4; ++c) {
      float s = 0.f;
      for (int j = 0; j < 4; ++j) s += Sn[r*4+j] * Ri[j*4+c];
      P[r*4+c] = s;
    }
  float* o = pp + t * 12;
  for (int r = 0; r < 3; ++r)
    for (int c = 0; c < 3; ++c) o[r*3+c] = P[r*4+c];   // rot, row-major
  o[9] = P[0*4+3]; o[10] = P[1*4+3]; o[11] = P[2*4+3]; // trans
}

// ---------------------------------------------------------------------------
// Kernel 1: homography warp + 3-view variance -> f16 volume [b,d,h,w][c]
// Channel-innermost layout: each position is a contiguous 64B record, written
// with 4x b128 stores, and DMA-friendly for the TDM row loads in k_conv.
// ---------------------------------------------------------------------------
__global__ __launch_bounds__(256) void k_volume(const float* __restrict__ feat,
                                                const float* __restrict__ dvals,
                                                const float* __restrict__ pp,
                                                _Float16* __restrict__ vol) {
  int idx = blockIdx.x * 256 + threadIdx.x;
  if (idx >= VOLN) return;
  int w = idx % WW;
  int h = (idx / WW) % HH;
  int b = idx / (WW * HH * DD);
  float depth = dvals[idx];

  int   cidx[2][4];
  float cwgt[2][4];
  for (int v = 0; v < 2; ++v) {
    const float* m = pp + (b*2 + v) * 12;
    float x = (float)w, y = (float)h;
    float X = m[0]*x + m[1]*y + m[2];
    float Y = m[3]*x + m[4]*y + m[5];
    float Z = m[6]*x + m[7]*y + m[8];
    X = X*depth + m[9]; Y = Y*depth + m[10]; Z = Z*depth + m[11];
    float px = X / Z, py = Y / Z;
    float fx = floorf(px), fy = floorf(py);
    int x0 = (int)fx, y0 = (int)fy;
    float tx = px - fx, ty = py - fy;
    int   xs[4] = { x0, x0+1, x0,   x0+1 };
    int   ys[4] = { y0, y0,   y0+1, y0+1 };
    float wl[4] = { (1.f-tx)*(1.f-ty), tx*(1.f-ty), (1.f-tx)*ty, tx*ty };
    for (int j = 0; j < 4; ++j) {
      bool valid = (xs[j] >= 0) && (xs[j] < WW) && (ys[j] >= 0) && (ys[j] < HH);
      int cx = min(max(xs[j], 0), WW-1);
      int cy = min(max(ys[j], 0), HH-1);
      cidx[v][j] = cy * WW + cx;
      cwgt[v][j] = valid ? wl[j] : 0.f;
    }
  }

  const float inv3 = 1.0f / 3.0f;
  size_t hw = (size_t)h * WW + w;
  _Float16 tmp[CC];
#pragma unroll
  for (int c = 0; c < CC; ++c) {
    float rv = feat[(((size_t)(b*VV + 0)*CC) + c) * HWN + hw];
    float s = rv, q = rv * rv;
    for (int v = 0; v < 2; ++v) {
      const float* fc = feat + (((size_t)(b*VV + 1 + v)*CC) + c) * HWN;
      float wa = cwgt[v][0]*fc[cidx[v][0]] + cwgt[v][1]*fc[cidx[v][1]] +
                 cwgt[v][2]*fc[cidx[v][2]] + cwgt[v][3]*fc[cidx[v][3]];
      s += wa; q += wa * wa;
    }
    float mean = s * inv3;
    float var  = q * inv3 - mean * mean;
    tmp[c] = (_Float16)var;
  }
  v8h* dst = (v8h*)(vol + (size_t)idx * CC);
#pragma unroll
  for (int g = 0; g < 4; ++g) {
    v8h t;
#pragma unroll
    for (int j = 0; j < 8; ++j) t[j] = tmp[g*8 + j];
    dst[g] = t;
  }
}

// ---------------------------------------------------------------------------
// TDM: DMA one contiguous row of WW*CC f16 elements (10KB) into LDS.
// D# built per CDNA5 ISA: group0 = {count=1, lds_addr,
// global_addr, type=2}; group1 = {data_size=2B, tensor_dim0=tile_dim0=5120,
// tensor_dim1=tile_dim1=1, dim0_stride=5120}. Groups 2/3 zero (<=2D tensor).
// ---------------------------------------------------------------------------
__device__ inline void tdm_load_row(const _Float16* gsrc, _Float16* ldst) {
#if USE_TDM
  unsigned long long ga = (unsigned long long)(uintptr_t)gsrc;
  unsigned lds_off = (unsigned)(uintptr_t)ldst;   // LDS offset = addr[31:0]
  const unsigned nelem = WW * CC;                 // 5120 f16
  u32x4 g0;
  g0[0] = 1u;                                      // count=1 (valid user D#)
  g0[1] = lds_off;                                 // lds_addr
  g0[2] = (unsigned)(ga & 0xffffffffu);            // global_addr[31:0]
  g0[3] = (unsigned)((ga >> 32) & 0x01ffffffu)     // global_addr[56:32]
        | (2u << 30);                              // type=2 ("image")
  i32x8 g1;
  g1[0] = (int)(1u << 16);                         // data_size=1 -> 2 bytes
  g1[1] = (int)((nelem & 0xffffu) << 16);          // tensor_dim0[15:0]
  g1[2] = (int)(((nelem >> 16) & 0xffffu)          // tensor_dim0[31:16]
        | (1u << 16));                             // tensor_dim1 = 1
  g1[3] = (int)((nelem & 0xffffu) << 16);          // tile_dim0 = 5120
  g1[4] = 1;                                       // tile_dim1 = 1, tile_dim2 = 0
  g1[5] = (int)nelem;                              // tensor_dim0_stride[31:0]
  g1[6] = 0;                                       // stride hi / dim1_stride lo
  g1[7] = 0;
  i32x4 gz = {0, 0, 0, 0};
#if defined(__clang_major__) && (__clang_major__ >= 23)
  i32x8 gz8 = {0, 0, 0, 0, 0, 0, 0, 0};
  __builtin_amdgcn_tensor_load_to_lds(g0, g1, gz, gz, gz8, 0);
#else
  __builtin_amdgcn_tensor_load_to_lds(g0, g1, gz, gz, 0);
#endif
  __builtin_amdgcn_s_wait_tensorcnt(0);
#else
  (void)gsrc; (void)ldst;
#endif
}

// ---------------------------------------------------------------------------
// Kernel 2: 3D conv via WMMA tap-decomposition.
//   partial[pos, tap] = sum_c vol[c,pos] * w[c,tap]   (16x16x32 f16 WMMA)
//   out[d,h,w] = sum_{kd,kh,kw} partial[(d+kd-1,h+kh-1,w+kw-1), tap] + bias
// One block = (b, TD x THT) tile over full W. Each extended (d',h') row is a
// contiguous 10KB block: TDM DMAs it into LDS (zeroed edges = free halo),
// 8 waves split the 11 W-segments, shift-add through LDS.
// ---------------------------------------------------------------------------
__global__ __launch_bounds__(256) void k_conv(const _Float16* __restrict__ vol,
                                              const float* __restrict__ wg,
                                              const float* __restrict__ bias,
                                              float* __restrict__ cost) {
  __shared__ float    P[176][32];        // tap partials for one extended row
  __shared__ float    acc[TD*THT*WW];    // output accumulators for the tile
  __shared__ _Float16 Avol[176*CC];      // staged row: index (w'+1)*CC + c

  int blk  = blockIdx.x;
  int hblk = blk % (HH/THT);
  int dblk = (blk / (HH/THT)) % (DD/TD);
  int b    = blk / ((HH/THT) * (DD/TD));
  int d0 = dblk * TD, h0 = hblk * THT;

  int lane = threadIdx.x & 31;
  int wid  = threadIdx.x >> 5;
  int hi   = lane >> 4;          // lane group (0: lanes 0-15, 1: lanes 16-31)
  int n    = lane & 15;          // WMMA column (tap) / row-in-group

  // B fragments: column n = tap, K = channel. contiguous K order used here (same as A):
  // lanes 0-15 hold K=0..15 (half j <-> K=j), lanes 16-31 hold K=16..31.
  v16h b0 = {}, b1 = {};
  for (int j = 0; j < 16; ++j) {
    int c = hi * 16 + j;
    b0[j] = (_Float16)wg[c*27 + n];
    b1[j] = (16 + n < 27) ? (_Float16)wg[c*27 + 16 + n] : (_Float16)0.0f;
  }

  for (int i = threadIdx.x; i < TD*THT*WW; i += 256) acc[i] = 0.f;
  // Zero whole staging row once: TDM rewrites only positions 1..160 each row,
  // so index 0 (w'=-1) and 161..175 stay zero -> halo padding for free.
  for (int i = threadIdx.x; i < 176*CC; i += 256) Avol[i] = (_Float16)0.0f;
  __syncthreads();

  for (int row = 0; row < (TD+2)*(THT+2); ++row) {
    int rd = row / (THT+2), rh = row % (THT+2);
    int dp = d0 + rd - 1, hp = h0 + rh - 1;
    bool rowok = (dp >= 0) && (dp < DD) && (hp >= 0) && (hp < HH);

    // --- stage phase: DMA the row into LDS ---
    const _Float16* gsrc = vol + (((size_t)b*DD + dp)*HH + hp) * (size_t)(WW*CC);
#if USE_TDM
    if (wid == 0 && rowok) tdm_load_row(gsrc, &Avol[CC]);
#else
    if (rowok)
      for (int i = threadIdx.x; i < (WW*CC)/8; i += 256)
        ((v8h*)&Avol[CC])[i] = ((const v8h*)gsrc)[i];
#endif
    __syncthreads();

    // --- compute phase: waves take W-segments round-robin (wave-uniform) ---
    for (int seg = wid; seg < 11; seg += 8) {
      int wi = seg * 16 + n;                     // Avol position index = w'+1
      // A fragment: row m = n (position), contiguous K order used here (same as B):
      //   halfs 0-15 <-> K = (hi?16:0)+0..15
      v16h a = {};
      if (rowok) {
        const _Float16* s = &Avol[wi*CC + (hi ? 16 : 0)];
        v8h lo  = *(const v8h*)(s);
        v8h hi8 = *(const v8h*)(s + 8);
        for (int j = 0; j < 8; ++j) { a[j] = lo[j]; a[8+j] = hi8[j]; }
      }
      v8f c0 = {}, c1 = {};
      c0 = __builtin_amdgcn_wmma_f32_16x16x32_f16(false, a, false, b0, (short)0, c0, false, false);
      c1 = __builtin_amdgcn_wmma_f32_16x16x32_f16(false, a, false, b1, (short)0, c1, false, false);
      // D layout: lane holds column N=n, rows M = r + 8*hi
      for (int r = 0; r < 8; ++r) {
        int m = r + (hi << 3);
        P[seg*16 + m][n]      = c0[r];   // taps 0..15
        P[seg*16 + m][16 + n] = c1[r];   // taps 16..26 (27..31 are zero-weight)
      }
    }
    __syncthreads();

    // --- accumulate phase: each (kd,kh) combo maps this row to one output row
    int ctd[9], cth[9], ctap[9]; int nc = 0;
    for (int kd = 0; kd < 3; ++kd) {
      int td = rd - kd; if (td < 0 || td >= TD) continue;
      for (int kh = 0; kh < 3; ++kh) {
        int th2 = rh - kh; if (th2 < 0 || th2 >= THT) continue;
        ctd[nc] = td; cth[nc] = th2; ctap[nc] = kd*9 + kh*3; ++nc;
      }
    }
    for (int i = threadIdx.x; i < nc * WW; i += 256) {
      int ci = i / WW, w = i % WW;
      int tb = ctap[ci];
      // buffer index = w+kw (position w-1+kw shifted by the -1 halo)
      float vs = P[w][tb] + P[w+1][tb+1] + P[w+2][tb+2];
      acc[(ctd[ci]*THT + cth[ci])*WW + w] += vs;   // unique index per i: no race
    }
    __syncthreads();   // also guards Avol/P reuse by the next row
  }

  float bv = bias[0];
  for (int i = threadIdx.x; i < TD*THT*WW; i += 256) {
    int td  = i / (THT*WW);
    int rem = i % (THT*WW);
    int th2 = rem / WW, w = rem % WW;
    cost[(((size_t)b*DD + d0 + td)*HH + h0 + th2)*WW + w] = acc[i] + bv;
  }
}

// ---------------------------------------------------------------------------
// Kernel 3: per-pixel softmax over D, depth regression, confidence, variance.
// Reads cost from the prob region of d_out and overwrites it in place.
// ---------------------------------------------------------------------------
__global__ __launch_bounds__(256) void k_post(float* __restrict__ prob,   // [B,D,H,W]
                                              const float* __restrict__ dvals,
                                              float* __restrict__ out_depth,
                                              float* __restrict__ out_conf,
                                              float* __restrict__ out_var) {
  int t = blockIdx.x * 256 + threadIdx.x;
  if (t >= BHW) return;
  int b = t / HWN;
  int hw = t % HWN;
  float* col = prob + (size_t)b * DD * HWN + hw;

  float p[DD], dv[DD];
  float mx = -3.402823e38f;
#pragma unroll
  for (int d = 0; d < DD; ++d) { p[d] = col[(size_t)d * HWN]; mx = fmaxf(mx, p[d]); }
  float s = 0.f;
#pragma unroll
  for (int d = 0; d < DD; ++d) { p[d] = __expf(p[d] - mx); s += p[d]; }
  float inv = 1.f / s;
  float depth = 0.f, didx = 0.f;
#pragma unroll
  for (int d = 0; d < DD; ++d) {
    p[d] *= inv;
    dv[d] = dvals[((size_t)b*DD + d) * HWN + hw];
    depth += p[d] * dv[d];
    didx  += p[d] * (float)d;
    col[(size_t)d * HWN] = p[d];
  }
  int di = min(max((int)didx, 0), DD - 1);
  float conf = 0.f;
  for (int j = di - 1; j <= di + 2; ++j)
    if (j >= 0 && j < DD) conf += p[j];
  float var = 0.f;
#pragma unroll
  for (int d = 0; d < DD; ++d) { float e = dv[d] - depth; var += p[d] * e * e; }

  out_depth[t] = depth;
  out_conf[t]  = conf;
  out_var[t]   = 3.f * sqrtf(var);
}

// ---------------------------------------------------------------------------
extern "C" void kernel_launch(void* const* d_in, const int* in_sizes, int n_in,
                              void* d_out, int out_size, void* d_ws, size_t ws_size,
                              hipStream_t stream) {
  (void)in_sizes; (void)n_in; (void)out_size; (void)ws_size;
  const float* feat  = (const float*)d_in[1];   // (B,V,C,H,W)
  const float* projm = (const float*)d_in[2];   // (B,V,2,4,4)
  const float* dvals = (const float*)d_in[3];   // (B,D,H,W)
  const float* reg_w = (const float*)d_in[5];   // (1,C,3,3,3)
  const float* reg_b = (const float*)d_in[6];   // (1,)

  float* out = (float*)d_out;
  float* out_depth = out;                        // B*H*W
  float* out_conf  = out + BHW;                  // B*H*W
  float* out_var   = out + 2*BHW;                // B*H*W
  float* out_prob  = out + 3*BHW;                // B*D*H*W (cost -> prob in place)

  float*     pp  = (float*)d_ws;                           // 48 floats proj params
  _Float16*  vol = (_Float16*)((char*)d_ws + 256);         // B*D*H*W*C halves (~126MB)

  k_setup<<<1, 32, 0, stream>>>(projm, pp);
  k_volume<<<(VOLN + 255)/256, 256, 0, stream>>>(feat, dvals, pp, vol);
  k_conv<<<BB * (DD/TD) * (HH/THT), 256, 0, stream>>>(vol, reg_w, reg_b, out_prob);
  k_post<<<(BHW + 255)/256, 256, 0, stream>>>(out_prob, dvals, out_depth, out_conf, out_var);
}
